// MultiBiGRU_51135880626535
// MI455X (gfx1250) — compile-verified, matrix operand order checked by reading, not run on hardware
//
#include <hip/hip_runtime.h>
#include <hip/hip_bf16.h>
#include <cstdint>
#include <cstddef>

// MultiBiGRU for MI455X (gfx1250, wave32).
//   - bf16 WMMA (v_wmma_f32_16x16x32_bf16), fp32 accumulate + fp32 gate math.
//   - W/U pre-transposed to [N][K] bf16: A and B fragments are two b128 loads.
//   - Recurrence: 1 workgroup (24 waves, 768 thr) per direction; h in LDS
//     (fp32 master + bf16 copy); rec in LDS; U hot in L2; per-step xw tile
//     (16x1152 f32) DMA'd into LDS via the Tensor Data Mover, double-buffered
//     one timestep ahead (TENSORcnt / s_wait_tensorcnt).
//   - Recurrent GEMM software-pipelines B-fragment loads one K-step ahead.

typedef __bf16 bf16_t;
typedef __attribute__((ext_vector_type(16))) __bf16 v16bf;
typedef __attribute__((ext_vector_type(8)))  float  v8f;
typedef __attribute__((ext_vector_type(4)))  unsigned int u32x4;
typedef __attribute__((ext_vector_type(8)))  int  i32x8;
typedef __attribute__((ext_vector_type(4)))  int  i32x4;

#define B_  16
#define T_  1024
#define H_  384
#define D_  768      // 2*H: input dim of every layer
#define G_  1152     // 3*H: gate dim
#define L_  2
#define BT_ (B_*T_)  // 16384
#define OUTC_ (4*H_) // 1536: columns of res / res_state
#define SCAN_THREADS 768   // 24 waves, 3 N-tiles per wave (72 tiles total)

__device__ __forceinline__ v8f wmma_bf16(v16bf a, v16bf b, v8f c) {
  return __builtin_amdgcn_wmma_f32_16x16x32_bf16(false, a, false, b,
                                                 (short)0, c, false, false);
}

union FragU { v16bf v; uint4 q[2]; };

// Load a 16x32 bf16 fragment from a row-major matrix (row stride in elements).
// ISA 7.12.2 16-bit A 16x32 layout: lanes 0-15 row=lane, K {0..7},{16..23};
// lanes 16-31 row=lane-16, K {8..15},{24..31}. Pre-transposed B uses the same
// pattern with row == N.
__device__ __forceinline__ v16bf load_frag(const bf16_t* base, int rowStride) {
  const int lane = threadIdx.x & 31;
  const int r    = lane & 15;
  const int koff = (lane < 16) ? 0 : 8;
  const bf16_t* p = base + (size_t)r * rowStride + koff;
  FragU f;
  f.q[0] = *(const uint4*)(p);
  f.q[1] = *(const uint4*)(p + 16);
  return f.v;
}

// ---------------------------------------------------------------------------
// TDM: DMA one [16 x 1152] f32 tile (row stride T_*G_ elements) into LDS.
// D# per ISA 8.3/8.4: group0 = {flags/count, lds_addr, global_addr, type=2},
// group1 = {data_size=4B, tensor_dim0=G, tensor_dim1=B, tile 1152x16, stride}.
// ---------------------------------------------------------------------------
__device__ __forceinline__ void tdm_load_tile(const float* gsrc, float* ldst) {
  const uint64_t ga = (uint64_t)(uintptr_t)gsrc;
  const uint32_t la = (uint32_t)(uintptr_t)ldst;   // LDS byte offset
  u32x4 g0 = { 1u,                                  // count=1, user descriptor
               la,                                  // lds_addr
               (uint32_t)ga,                        // global_addr[31:0]
               ((uint32_t)(ga >> 32) & 0x01FFFFFFu) | (2u << 30) }; // [56:32], type=2
  i32x8 g1 = { (int)(2u << 16),                     // data_size = 4 bytes
               (int)((uint32_t)G_ << 16),           // tensor_dim0[15:0]=1152
               (int)((uint32_t)B_ << 16),           // dim0 hi=0 | tensor_dim1[15:0]=16
               (int)((uint32_t)G_ << 16),           // dim1 hi=0 | tile_dim0=1152
               (int)B_,                             // tile_dim1=16, tile_dim2=0
               (int)(T_ * G_),                      // tensor_dim0_stride[31:0]
               0, 0 };
  i32x4 gz = { 0, 0, 0, 0 };
#if __clang_major__ >= 23
  i32x8 gz8 = { 0, 0, 0, 0, 0, 0, 0, 0 };
  __builtin_amdgcn_tensor_load_to_lds(g0, g1, gz, gz, gz8, 0);
#else
  __builtin_amdgcn_tensor_load_to_lds(g0, g1, gz, gz, 0);
#endif
}

// ---------------------------------------------------------------------------
// Conversion kernels
// ---------------------------------------------------------------------------
__global__ void k_cvt_bf16(const float* __restrict__ src, bf16_t* __restrict__ dst,
                           int rows, int cols, int srcStride) {
  int idx = blockIdx.x * blockDim.x + threadIdx.x;
  if (idx >= rows * cols) return;
  int r = idx / cols, c = idx - r * cols;
  dst[(size_t)r * cols + c] = (bf16_t)src[(size_t)r * srcStride + c];
}

// src [K][N] fp32 -> dst [N][K] bf16
__global__ void k_transpose_cvt(const float* __restrict__ src, bf16_t* __restrict__ dst,
                                int K, int N) {
  int idx = blockIdx.x * blockDim.x + threadIdx.x;
  if (idx >= K * N) return;
  int k = idx / N, n = idx - k * N;
  dst[(size_t)n * K + k] = (bf16_t)src[(size_t)k * N + n];
}

// ---------------------------------------------------------------------------
// Input GEMM: xw[dir][b*T+t][0:1152] = Xb @ W[dir] + b0[dir]
// M=16384, N=1152, K=768.  1 wave = 16(M) x 64(N), K-loop in steps of 32.
// HBM-bound; 2304x2 blocks provide latency hiding.
// ---------------------------------------------------------------------------
__global__ void k_xw_gemm(const bf16_t* __restrict__ Xb,   // [BT][D]
                          const bf16_t* __restrict__ WtL,  // [2][G][D] (transposed)
                          const float*  __restrict__ bL,   // [2][2][G]
                          float* __restrict__ xw) {        // [2][BT][G]
  const int dir   = blockIdx.y;
  const int wave  = threadIdx.x >> 5;
  const int unit  = blockIdx.x * 8 + wave;    // 1024 mtiles * 18 nchunks
  const int mtile = unit / 18;
  const int n0    = (unit - mtile * 18) * 64;

  const bf16_t* Wd   = WtL + (size_t)dir * G_ * D_;
  const float*  bIn  = bL + (size_t)dir * 2 * G_;   // b[l][dir][0][:]
  const bf16_t* arow = Xb + (size_t)mtile * 16 * D_;

  v8f acc[4] = {};
  for (int kb = 0; kb < D_; kb += 32) {
    v16bf a  = load_frag(arow + kb, D_);
    v16bf b0 = load_frag(Wd + (size_t)(n0 +  0) * D_ + kb, D_);
    v16bf b1 = load_frag(Wd + (size_t)(n0 + 16) * D_ + kb, D_);
    v16bf b2 = load_frag(Wd + (size_t)(n0 + 32) * D_ + kb, D_);
    v16bf b3 = load_frag(Wd + (size_t)(n0 + 48) * D_ + kb, D_);
    acc[0] = wmma_bf16(a, b0, acc[0]);
    acc[1] = wmma_bf16(a, b1, acc[1]);
    acc[2] = wmma_bf16(a, b2, acc[2]);
    acc[3] = wmma_bf16(a, b3, acc[3]);
  }

  const int lane = threadIdx.x & 31;
  const int nl   = lane & 15;
  const int m0   = (lane < 16) ? 0 : 8;
  float* xwd = xw + (size_t)dir * BT_ * G_;
  #pragma unroll
  for (int j = 0; j < 4; ++j) {
    const int n = n0 + j * 16 + nl;
    const float bv = bIn[n];
    #pragma unroll
    for (int r = 0; r < 8; ++r)
      xwd[(size_t)(mtile * 16 + m0 + r) * G_ + n] = acc[j][r] + bv;
  }
}

// ---------------------------------------------------------------------------
// Recurrent scan: 1 workgroup / direction, 24 waves, T sequential steps.
// LDS layout: rec[16x1152]f32 | xw0[16x1152]f32 | xw1[16x1152]f32 |
//             hF[16x384]f32 | hB[16x384]bf16   (total 258048 B)
// ---------------------------------------------------------------------------
__global__ void k_gru_scan(const float*  __restrict__ xw,   // [2][BT][G]
                           const bf16_t* __restrict__ UtL,  // [2][G][H] (transposed)
                           const float*  __restrict__ bL,   // [2][2][G]
                           float* __restrict__ res,         // [BT][1536]
                           float* __restrict__ state,       // [B][1536]
                           int layer) {
  extern __shared__ char smem[];
  float*  recL = (float*)smem;                       // 16*1152
  float*  xwT[2];
  xwT[0] = recL + B_ * G_;                           // 16*1152
  xwT[1] = xwT[0] + B_ * G_;                         // 16*1152
  float*  hF = xwT[1] + B_ * G_;                     // 16*384
  bf16_t* hB = (bf16_t*)(hF + B_ * H_);              // 16*384

  const int dir  = blockIdx.x;
  const int tid  = threadIdx.x;
  const int wave = tid >> 5;
  const int lane = tid & 31;

  const bf16_t* Ud   = UtL + (size_t)dir * G_ * H_;
  const float*  bRec = bL + (size_t)dir * 2 * G_ + G_;   // b[l][dir][1][:]
  const float*  xwd  = xw + (size_t)dir * BT_ * G_;
  const int outCol = layer * 2 * H_ + dir * H_;

  for (int i = tid; i < B_ * H_; i += blockDim.x) { hF[i] = 0.f; hB[i] = (bf16_t)0.f; }
  __syncthreads();

  // Prime the pipeline: DMA xw tile for the first timestep into buffer 0.
  const int t0 = dir ? (T_ - 1) : 0;
  if (wave == 0) tdm_load_tile(xwd + (size_t)t0 * G_, xwT[0]);

  for (int ti = 0; ti < T_; ++ti) {
    const int t   = dir ? (T_ - 1 - ti) : ti;
    const int buf = ti & 1;

    // ---- rec = h @ U : wave handles N-tiles [wave*3, wave*3+3) ----
    v16bf aF[12];
    #pragma unroll
    for (int kb = 0; kb < 12; ++kb) aF[kb] = load_frag(hB + kb * 32, H_);

    #pragma unroll
    for (int i = 0; i < 3; ++i) {
      const int n0 = (wave * 3 + i) * 16;
      const bf16_t* ub = Ud + (size_t)n0 * H_;
      v8f acc = {};
      v16bf bcur = load_frag(ub, H_);
      #pragma unroll
      for (int kb = 0; kb < 12; ++kb) {
        v16bf bnxt = bcur;
        if (kb < 11) bnxt = load_frag(ub + (kb + 1) * 32, H_);  // prefetch next K-step
        acc = wmma_bf16(aF[kb], bcur, acc);
        bcur = bnxt;
      }
      const int nl = n0 + (lane & 15);
      const int m0 = (lane < 16) ? 0 : 8;
      #pragma unroll
      for (int r = 0; r < 8; ++r)
        recL[(m0 + r) * G_ + nl] = acc[r];
    }

    // ---- wave 0: launch next tile's DMA, then ensure current tile landed ----
    if (wave == 0) {
      if (ti + 1 < T_) {
        const int tn = dir ? (t - 1) : (t + 1);
        tdm_load_tile(xwd + (size_t)tn * G_, xwT[buf ^ 1]);
        __builtin_amdgcn_s_wait_tensorcnt(1);   // current tile complete
      } else {
        __builtin_amdgcn_s_wait_tensorcnt(0);
      }
    }
    __syncthreads();

    // ---- gates + h update: 16*384 elements / 768 threads = 8 each ----
    const float* xwL = xwT[buf];
    #pragma unroll
    for (int k = 0; k < (B_ * H_) / SCAN_THREADS; ++k) {
      const int idx = tid + k * SCAN_THREADS;
      const int m = idx / H_;
      const int j = idx - m * H_;
      const float xz = xwL[m * G_ + j];
      const float xr = xwL[m * G_ + H_ + j];
      const float xh = xwL[m * G_ + 2 * H_ + j];
      const float rz = recL[m * G_ + j]          + bRec[j];
      const float rr = recL[m * G_ + H_ + j]     + bRec[H_ + j];
      const float rh = recL[m * G_ + 2 * H_ + j] + bRec[2 * H_ + j];
      const float z  = 1.f / (1.f + __expf(-(xz + rz)));
      const float r  = 1.f / (1.f + __expf(-(xr + rr)));
      const float hh = tanhf(xh + r * rh);
      const float hn = z * hF[m * H_ + j] + (1.f - z) * hh;
      hF[m * H_ + j] = hn;
      hB[m * H_ + j] = (bf16_t)hn;
      res[(size_t)(m * T_ + t) * OUTC_ + outCol + j] = hn;
    }
    __syncthreads();
  }

  for (int idx = tid; idx < B_ * H_; idx += blockDim.x) {
    const int m = idx / H_, j = idx - m * H_;
    state[(size_t)m * OUTC_ + outCol + j] = hF[idx];
  }
}

// ---------------------------------------------------------------------------
// Host orchestration
// ---------------------------------------------------------------------------
extern "C" void kernel_launch(void* const* d_in, const int* in_sizes, int n_in,
                              void* d_out, int out_size, void* d_ws, size_t ws_size,
                              hipStream_t stream) {
  const float* x = (const float*)d_in[0];   // [16,1024,768]
  const float* W = (const float*)d_in[1];   // [2,2,768,1152]
  const float* U = (const float*)d_in[2];   // [2,2,384,1152]
  const float* b = (const float*)d_in[3];   // [2,2,2,1152]

  float* out   = (float*)d_out;
  float* state = out;                          // res_state: [16,1536]
  float* res   = out + (size_t)B_ * OUTC_;     // res:       [16,1024,1536]

  // Workspace: Wt 7.08MB | Ut 3.54MB | Xb 25.2MB | xw 151MB
  char* ws = (char*)d_ws;
  bf16_t* Wt = (bf16_t*)ws;                                          // [2][2][G][D]
  bf16_t* Ut = (bf16_t*)(ws + (size_t)4 * G_ * D_ * 2);              // [2][2][G][H]
  bf16_t* Xb = (bf16_t*)(ws + (size_t)4 * G_ * D_ * 2
                            + (size_t)4 * G_ * H_ * 2);              // [BT][D]
  float* xwbuf = (float*)(ws + (size_t)4 * G_ * D_ * 2
                             + (size_t)4 * G_ * H_ * 2
                             + (size_t)BT_ * D_ * 2);                // [2][BT][G]

  const int thr = 256;
  for (int l = 0; l < L_; ++l)
    for (int d = 0; d < 2; ++d) {
      const int s = l * 2 + d;
      k_transpose_cvt<<<(D_ * G_ + thr - 1) / thr, thr, 0, stream>>>(
          W + (size_t)s * D_ * G_, Wt + (size_t)s * G_ * D_, D_, G_);
      k_transpose_cvt<<<(H_ * G_ + thr - 1) / thr, thr, 0, stream>>>(
          U + (size_t)s * H_ * G_, Ut + (size_t)s * G_ * H_, H_, G_);
    }

  // rec + 2 xw tiles + hF (f32) + hB (bf16)
  const size_t ldsBytes = (size_t)B_ * G_ * 4 * 3 + (size_t)B_ * H_ * 4 + (size_t)B_ * H_ * 2;

  for (int l = 0; l < L_; ++l) {
    if (l == 0)
      k_cvt_bf16<<<(BT_ * D_ + thr - 1) / thr, thr, 0, stream>>>(x, Xb, BT_, D_, D_);
    else  // layer-1 input = layer-0 output, cols [0,768) of res (stride 1536)
      k_cvt_bf16<<<(BT_ * D_ + thr - 1) / thr, thr, 0, stream>>>(res, Xb, BT_, D_, OUTC_);

    dim3 g(1024 * 18 / 8, 2);  // 2304 blocks x 2 directions
    k_xw_gemm<<<g, thr, 0, stream>>>(Xb, Wt + (size_t)l * 2 * G_ * D_,
                                     b + (size_t)l * 2 * 2 * G_, xwbuf);

    k_gru_scan<<<2, SCAN_THREADS, ldsBytes, stream>>>(
        xwbuf, Ut + (size_t)l * 2 * G_ * H_, b + (size_t)l * 2 * 2 * G_,
        res, state, l);
  }
}
